// VQEncoder_53068615909687
// MI455X (gfx1250) — matrix operand order repaired and checked by reading, not compiled, not run on hardware
//
#include <hip/hip_runtime.h>
#include <hip/hip_bf16.h>
#include <cstdint>

#ifndef TRY_ASYNC
#define TRY_ASYNC 1            // gfx1250 GLOBAL_LOAD_ASYNC_TO_LDS staging
#endif

// ---------------- problem constants ----------------
#define D_MODEL   1024
#define NUM_CODES 8192
#define NROWS     16384          // 8 * 2048
#define MT        128            // M tile per workgroup
#define NT        128            // N tile per workgroup
#define KC        64             // K chunk per LDS stage (2 x 32-k sub-blocks)
#define NCHUNK    (D_MODEL / KC) // 16
#define NSPLIT    4              // N splits across workgroups
#define NB_PER_SPLIT ((NUM_CODES / NSPLIT) / NT)   // 16
#define LDSTRIDE  40             // uint16 per 32-k LDS row (64B data + 16B pad)

typedef __attribute__((ext_vector_type(8)))  float  v8f;
typedef __attribute__((ext_vector_type(16))) __bf16 v16bf;

union V16 { uint4 q[2]; v16bf v; };

__device__ __forceinline__ uint16_t f2bf(float f) {
    uint32_t u = __float_as_uint(f);
    uint32_t r = u + 0x7FFFu + ((u >> 16) & 1u);   // round-to-nearest-even
    return (uint16_t)(r >> 16);
}

__device__ __forceinline__ unsigned orderable(float v) {
    unsigned u = __float_as_uint(v);
    return (u & 0x80000000u) ? ~u : (u | 0x80000000u);
}

// 16B global -> LDS segment copy. Async DMA (ASYNCcnt) when available.
__device__ __forceinline__ void seg_ld16(uint16_t* lds, const uint16_t* g) {
#if TRY_ASYNC
    uint32_t l = (uint32_t)(uintptr_t)lds;          // LDS byte offset (addr[31:0])
    uint64_t a = (uint64_t)(uintptr_t)g;
    asm volatile("global_load_async_to_lds_b128 %0, %1, off"
                 :: "v"(l), "v"(a) : "memory");
#else
    *(uint4*)lds = *(const uint4*)g;
#endif
}

__device__ __forceinline__ void seg_wait() {
#if TRY_ASYNC
    asm volatile("s_wait_asynccnt 0x0" ::: "memory");
#endif
}

// ---- codebook: f32 -> bf16, and exact fp32 squared norms -------------------
__global__ __launch_bounds__(256) void vq_prep_cb(const float* __restrict__ cb,
                                                  uint16_t* __restrict__ cb16,
                                                  float* __restrict__ cnorm) {
    const int row = blockIdx.x;
    const float* ci = cb + (size_t)row * D_MODEL;
    uint16_t* co = cb16 + (size_t)row * D_MODEL;
    float s = 0.f;
    for (int j = threadIdx.x; j < D_MODEL; j += 256) {
        float f = ci[j];
        s = fmaf(f, f, s);
        co[j] = f2bf(f);
    }
    __shared__ float red[8];
    for (int m = 16; m >= 1; m >>= 1) s += __shfl_xor(s, m, 32);
    if ((threadIdx.x & 31) == 0) red[threadIdx.x >> 5] = s;
    __syncthreads();
    if (threadIdx.x == 0) {
        float t = 0.f;
        for (int w = 0; w < 8; ++w) t += red[w];
        cnorm[row] = t;
    }
}

// ---- z: f32 -> bf16 (one pass, removes redundant per-tile conversion) -----
__global__ __launch_bounds__(256) void vq_prep_z(const float* __restrict__ z,
                                                 uint16_t* __restrict__ zb) {
    size_t i = ((size_t)blockIdx.x * 256 + threadIdx.x) * 8;
    const float4* g = (const float4*)(z + i);
    float4 f0 = g[0], f1 = g[1];
    uint32_t d0 = ((uint32_t)f2bf(f0.y) << 16) | f2bf(f0.x);
    uint32_t d1 = ((uint32_t)f2bf(f0.w) << 16) | f2bf(f0.z);
    uint32_t d2 = ((uint32_t)f2bf(f1.y) << 16) | f2bf(f1.x);
    uint32_t d3 = ((uint32_t)f2bf(f1.w) << 16) | f2bf(f1.z);
    *(uint4*)(zb + i) = make_uint4(d0, d1, d2, d3);
}

// ---- init packed argmin keys to +inf -------------------------------------
__global__ void vq_init(unsigned long long* __restrict__ keys) {
    int i = blockIdx.x * 256 + threadIdx.x;
    if (i < NROWS) keys[i] = 0xFFFFFFFFFFFFFFFFULL;
}

// ---- fused bf16-WMMA GEMM + row argmin -----------------------------------
template <bool PRECONV>
__global__ __launch_bounds__(256) void vq_main(const float* __restrict__ z,
                                               const uint16_t* __restrict__ zb,
                                               const uint16_t* __restrict__ cb16,
                                               const float* __restrict__ cnorm,
                                               unsigned long long* __restrict__ keys) {
    __shared__ uint16_t lA[2][2 * MT * LDSTRIDE];   // [buf][kc2*128 + row][40]
    __shared__ uint16_t lB[2][2 * NT * LDSTRIDE];

    const int tid  = threadIdx.x;
    const int lane = tid & 31;
    const int wave = tid >> 5;
    const int wm   = wave & 3;      // 4 waves along M  -> 32 rows each
    const int wn   = wave >> 2;     // 2 waves along N  -> 64 cols each
    const int lrow = lane & 15;
    const int mb    = blockIdx.x >> 2;   // 0..127
    const int split = blockIdx.x & 3;    // 0..3
    const int mRowBase   = mb * MT;
    const int nSplitBase = split * (NUM_CODES / NSPLIT);

    float    bestV[2][8];
    unsigned bestN[2][8];
#pragma unroll
    for (int mi = 0; mi < 2; ++mi)
#pragma unroll
        for (int k = 0; k < 8; ++k) { bestV[mi][k] = 3.4e38f; bestN[mi][k] = 0u; }

    const int aoff = (lane < 16) ? 0 : 8;    // A: 16B K-group select
    const int boff = (lane < 16) ? 0 : 8;    // B: 16B K-group select

    for (int nb = 0; nb < NB_PER_SPLIT; ++nb) {
        const int nBase = nSplitBase + nb * NT;

        v8f acc[2][4];
#pragma unroll
        for (int mi = 0; mi < 2; ++mi)
#pragma unroll
            for (int ni = 0; ni < 4; ++ni) { v8f zv = {}; acc[mi][ni] = zv; }

        float cn[4];
#pragma unroll
        for (int ni = 0; ni < 4; ++ni)
            cn[ni] = cnorm[nBase + wn * 64 + ni * 16 + lrow];

        // stage one KC=64 chunk: 1024 16B segments each for A and B
        auto stage = [&](int buf, int kc) {
#pragma unroll
            for (int t = 0; t < 4; ++t) {
                int s = tid + t * 256;               // 0..1023
                int r = s >> 3, q = s & 7;
                int kc2 = q >> 2, p = q & 3;
                uint16_t* l = &lA[buf][(kc2 * MT + r) * LDSTRIDE + p * 8];
                if (PRECONV) {
                    seg_ld16(l, zb + (size_t)(mRowBase + r) * D_MODEL + kc + kc2 * 32 + p * 8);
                } else {
                    const float4* g = (const float4*)(z + (size_t)(mRowBase + r) * D_MODEL
                                                        + kc + kc2 * 32 + p * 8);
                    float4 f0 = g[0], f1 = g[1];
                    uint32_t d0 = ((uint32_t)f2bf(f0.y) << 16) | f2bf(f0.x);
                    uint32_t d1 = ((uint32_t)f2bf(f0.w) << 16) | f2bf(f0.z);
                    uint32_t d2 = ((uint32_t)f2bf(f1.y) << 16) | f2bf(f1.x);
                    uint32_t d3 = ((uint32_t)f2bf(f1.w) << 16) | f2bf(f1.z);
                    *(uint4*)l = make_uint4(d0, d1, d2, d3);
                }
            }
#pragma unroll
            for (int t = 0; t < 4; ++t) {
                int s = tid + t * 256;
                int r = s >> 3, q = s & 7;
                int kc2 = q >> 2, p = q & 3;
                seg_ld16(&lB[buf][(kc2 * NT + r) * LDSTRIDE + p * 8],
                         cb16 + (size_t)(nBase + r) * D_MODEL + kc + kc2 * 32 + p * 8);
            }
        };

        stage(0, 0);
        seg_wait();
        __syncthreads();

        for (int kci = 0; kci < NCHUNK; ++kci) {
            const int cur = kci & 1;
            if (kci + 1 < NCHUNK) stage(cur ^ 1, (kci + 1) * KC);   // async overlaps WMMAs

#pragma unroll
            for (int kc2 = 0; kc2 < 2; ++kc2) {
                V16 av[2], bv[4];
#pragma unroll
                for (int mi = 0; mi < 2; ++mi) {
                    const uint16_t* pa =
                        &lA[cur][(kc2 * MT + wm * 32 + mi * 16 + lrow) * LDSTRIDE + aoff];
                    av[mi].q[0] = *(const uint4*)pa;          // K 0..7  / 8..15
                    av[mi].q[1] = *(const uint4*)(pa + 16);   // K 16..23 / 24..31
                }
#pragma unroll
                for (int ni = 0; ni < 4; ++ni) {
                    const uint16_t* pb =
                        &lB[cur][(kc2 * NT + wn * 64 + ni * 16 + lrow) * LDSTRIDE + boff];
                    bv[ni].q[0] = *(const uint4*)pb;
                    bv[ni].q[1] = *(const uint4*)(pb + 16);
                }
#pragma unroll
                for (int mi = 0; mi < 2; ++mi)
#pragma unroll
                    for (int ni = 0; ni < 4; ++ni)
                        acc[mi][ni] = __builtin_amdgcn_wmma_f32_16x16x32_bf16(
                            false, av[mi].v, false, bv[ni].v,
                            (short)0, acc[mi][ni], false, false);
            }
            seg_wait();          // fence only the next-buffer DMA this wave issued
            __syncthreads();
        }

        // epilogue: score = ||c||^2 - 2 z.c ; running per-lane argmin
#pragma unroll
        for (int mi = 0; mi < 2; ++mi)
#pragma unroll
            for (int k = 0; k < 8; ++k)
#pragma unroll
                for (int ni = 0; ni < 4; ++ni) {
                    float s = fmaf(-2.f, acc[mi][ni][k], cn[ni]);
                    if (s < bestV[mi][k]) {
                        bestV[mi][k] = s;
                        bestN[mi][k] = (unsigned)(nBase + wn * 64 + ni * 16 + lrow);
                    }
                }
    }

    // cross-lane (N direction) reduce within 16-lane groups, then atomic combine
#pragma unroll
    for (int mi = 0; mi < 2; ++mi)
#pragma unroll
        for (int k = 0; k < 8; ++k) {
            float v = bestV[mi][k];
            unsigned n = bestN[mi][k];
#pragma unroll
            for (int m = 8; m >= 1; m >>= 1) {
                float ov = __shfl_xor(v, m, 32);
                unsigned on = (unsigned)__shfl_xor((int)n, m, 32);
                if (ov < v || (ov == v && on < n)) { v = ov; n = on; }
            }
            if (lrow == 0) {
                int row = mRowBase + wm * 32 + mi * 16 + k + ((lane >> 4) << 3);
                unsigned long long key =
                    ((unsigned long long)orderable(v) << 32) | (unsigned long long)n;
                atomicMin(&keys[row], key);
            }
        }
}

// ---- gather quantized rows (fp32), per-row loss, indices ------------------
__global__ __launch_bounds__(256) void vq_finalize(const float* __restrict__ z,
                                                   const float* __restrict__ cb,
                                                   const unsigned long long* __restrict__ keys,
                                                   float* __restrict__ outq,
                                                   float* __restrict__ outidx,
                                                   float* __restrict__ rowloss) {
    const int row = blockIdx.x;
    const unsigned idx = (unsigned)(keys[row] & 0xFFFFFFFFULL);
    const float* c  = cb + (size_t)idx * D_MODEL;
    const float* zr = z  + (size_t)row * D_MODEL;
    float* o = outq + (size_t)row * D_MODEL;
    float s = 0.f;
    for (int j = threadIdx.x; j < D_MODEL; j += 256) {
        float q = c[j];
        o[j] = q;
        float d = zr[j] - q;
        s = fmaf(d, d, s);
    }
    __shared__ float red[8];
    for (int m = 16; m >= 1; m >>= 1) s += __shfl_xor(s, m, 32);
    if ((threadIdx.x & 31) == 0) red[threadIdx.x >> 5] = s;
    __syncthreads();
    if (threadIdx.x == 0) {
        float t = 0.f;
        for (int w = 0; w < 8; ++w) t += red[w];
        rowloss[row] = t;
        outidx[row] = (float)idx;
    }
}

// ---- deterministic scalar loss = 1.25 * mean((z-q)^2) ---------------------
__global__ __launch_bounds__(256) void vq_loss(const float* __restrict__ rowloss,
                                               float* __restrict__ outloss) {
    float s = 0.f;
    for (int i = threadIdx.x; i < NROWS; i += 256) s += rowloss[i];
    __shared__ float red[8];
    for (int m = 16; m >= 1; m >>= 1) s += __shfl_xor(s, m, 32);
    if ((threadIdx.x & 31) == 0) red[threadIdx.x >> 5] = s;
    __syncthreads();
    if (threadIdx.x == 0) {
        float t = 0.f;
        for (int w = 0; w < 8; ++w) t += red[w];
        outloss[0] = t * (1.25f / ((float)NROWS * (float)D_MODEL));
    }
}

// ---------------------------------------------------------------------------
extern "C" void kernel_launch(void* const* d_in, const int* in_sizes, int n_in,
                              void* d_out, int out_size, void* d_ws, size_t ws_size,
                              hipStream_t stream) {
    const float* z  = (const float*)d_in[0];   // [16384, 1024] f32
    const float* cb = (const float*)d_in[1];   // [8192, 1024]  f32

    char* ws = (char*)d_ws;
    uint16_t* cb16 = (uint16_t*)ws;                                        // 16 MB
    float*    cnorm = (float*)(ws + (size_t)16777216);                     // 32 KB
    unsigned long long* keys = (unsigned long long*)(ws + (size_t)16809984); // 128 KB
    float*    rowloss = (float*)(ws + (size_t)16941056);                   // 64 KB
    uint16_t* zb16 = (uint16_t*)(ws + (size_t)17006592);                   // 32 MB (optional)

    float* outq    = (float*)d_out;            // [16384*1024]
    float* outloss = outq + (size_t)16777216;  // [1]
    float* outidx  = outloss + 1;              // [16384] (as float)

    const bool preconv = ws_size >= (size_t)50561024;

    vq_prep_cb<<<NUM_CODES, 256, 0, stream>>>(cb, cb16, cnorm);
    vq_init<<<(NROWS + 255) / 256, 256, 0, stream>>>(keys);
    if (preconv) {
        vq_prep_z<<<(NROWS * D_MODEL / 8) / 256, 256, 0, stream>>>(z, zb16);
        vq_main<true><<<(NROWS / MT) * NSPLIT, 256, 0, stream>>>(z, zb16, cb16, cnorm, keys);
    } else {
        vq_main<false><<<(NROWS / MT) * NSPLIT, 256, 0, stream>>>(z, nullptr, cb16, cnorm, keys);
    }
    vq_finalize<<<NROWS, 256, 0, stream>>>(z, cb, keys, outq, outidx, rowloss);
    vq_loss<<<1, 256, 0, stream>>>(rowloss, outloss);
}
